// TensoricSwiGLU_455266534074
// MI455X (gfx1250) — matrix operand order repaired and checked by reading, not compiled, not run on hardware
//
#include <hip/hip_runtime.h>
#include <hip/hip_bf16.h>

// CDNA5 (gfx1250) wave32 WMMA types
typedef __attribute__((ext_vector_type(16))) __bf16 v16bf;
typedef __attribute__((ext_vector_type(8)))  float  v8f;
typedef __attribute__((ext_vector_type(4)))  int    v4i;

// Problem constants (match reference)
constexpr int Bdim  = 4;
constexpr int Sdim  = 2048;
constexpr int Hdim  = 512;     // hidden
constexpr int Idim  = 1024;    // intermediate
constexpr int Edim  = 8;       // experts
constexpr int TOPK  = 2;
constexpr int TWOI  = 2 * Idim;

constexpr int NTOK  = Bdim * Sdim;     // 8192 tokens
constexpr int NPAIR = NTOK * TOPK;     // 16384 (token,expert) pairs

constexpr int TILE_M    = 32;                  // tokens per block
constexpr int MAX_TILES = NPAIR / TILE_M;      // 512
constexpr int IC        = 32;                  // intermediate chunk
constexpr int NKB       = Hdim / 32;           // 16 K-blocks (K=32 per WMMA)
constexpr int NCH       = Idim / IC;           // 32 chunks

// packed-weight sizes (bf16, fragment-ordered)
constexpr int W13P_PER_CHUNK = NKB * 4 * 32 * 16;   // 32768 elems = 64 KB
constexpr int W2P_PER_CHUNK  = 32 * 32 * 16;        // 16384 elems = 32 KB
constexpr int W13P_TOTAL = Edim * NCH * W13P_PER_CHUNK;  // 8,388,608 (= E*H*2I)
constexpr int W2P_TOTAL  = Edim * NCH * W2P_PER_CHUNK;   // 4,194,304 (= E*I*H)

// --- fragment index helpers (this file's contiguous K order, shared by A and B) ---
// A 16x32 bf16: lane = (M%16) + 16*(K>>4), elem j = K&15
__device__ __forceinline__ int a_half(int kk) { return (kk >> 4) & 1; }
__device__ __forceinline__ int a_j(int kk)    { return kk & 15; }
// B 32x16 bf16: lane = N + 16*(K>>4), elem j = K&15

// --- gfx1250 async global->LDS DMA (guarded; sync fallback) ---
#if defined(__has_builtin)
#if __has_builtin(__builtin_amdgcn_global_load_async_to_lds_b128)
#define HAVE_ASYNC_LDS 1
#endif
#if __has_builtin(__builtin_amdgcn_s_wait_asynccnt)
#define WAIT_ASYNC(n) __builtin_amdgcn_s_wait_asynccnt(n)
#endif
#endif
#ifndef WAIT_ASYNC
#define WAIT_ASYNC(n) asm volatile("s_wait_asynccnt " #n ::: "memory")
#endif

// builtin signature (from hipcc diagnostic): (AS1 v4i*, AS3 v4i*, imm int, imm int)
typedef __attribute__((address_space(1))) v4i* gptr_v4i;
typedef __attribute__((address_space(3))) v4i* lptr_v4i;

__device__ __forceinline__ void cp16_to_lds(void* lds, const void* g) {
#if defined(HAVE_ASYNC_LDS)
  __builtin_amdgcn_global_load_async_to_lds_b128((gptr_v4i)(g), (lptr_v4i)(lds), 0, 0);
#else
  *(int4*)lds = *(const int4*)g;
#endif
}

// ---------------- routing kernels ----------------
extern "C" __global__ void moe_zero_cnt(int* __restrict__ cnt) {
  if (threadIdx.x < Edim) cnt[threadIdx.x] = 0;
}

extern "C" __global__ void moe_scatter(const int* __restrict__ eidx,
                                       int* __restrict__ cnt,
                                       int* __restrict__ buckets) {
  int p = blockIdx.x * blockDim.x + threadIdx.x;
  if (p < NPAIR) {
    int e = eidx[p];
    int pos = atomicAdd(&cnt[e], 1);
    buckets[e * NPAIR + pos] = p;
  }
}

// ---------------- one-time weight pack: fp32 -> bf16 fragment order ----------------
// dest d = ((((e*NCH + ci)*NKB + kb)*4 + sel)*32 + lane)*16 + j
extern "C" __global__ void moe_pack13(const float* __restrict__ w13,
                                      __bf16* __restrict__ w13P) {
  #pragma unroll
  for (int i = 0; i < 8; ++i) {
    int d = (blockIdx.x * 8 + i) * 256 + threadIdx.x;
    int j    = d & 15;
    int lane = (d >> 4) & 31;
    int sel  = (d >> 9) & 3;
    int kb   = (d >> 11) & 15;
    int ci   = (d >> 15) & 31;
    int e    = d >> 20;
    int n  = lane & 15;
    int kk = ((lane >> 4) << 4) + j;       // B-fragment K
    int r  = (kb << 5) + kk;               // H row
    int cc = (sel << 4) + n;               // 0..31 gate, 32..63 up
    int col = (cc < 32) ? ((ci << 5) + cc) : (Idim + (ci << 5) + (cc - 32));
    w13P[d] = (__bf16)w13[((size_t)e * Hdim + r) * TWOI + col];
  }
}

// dest d = (((e*NCH + ci)*32 + nt)*32 + lane)*16 + j
extern "C" __global__ void moe_pack2(const float* __restrict__ w2,
                                     __bf16* __restrict__ w2P) {
  #pragma unroll
  for (int i = 0; i < 8; ++i) {
    int d = (blockIdx.x * 8 + i) * 256 + threadIdx.x;
    int j    = d & 15;
    int lane = (d >> 4) & 31;
    int nt   = (d >> 9) & 31;
    int ci   = (d >> 14) & 31;
    int e    = d >> 19;
    int n  = lane & 15;
    int kk = ((lane >> 4) << 4) + j;       // chunk row (K)
    w2P[d] = (__bf16)w2[((size_t)e * Idim + (ci << 5) + kk) * Hdim + (nt << 4) + n];
  }
}

// ---------------- grouped GEMM + SwiGLU ----------------
extern "C" __global__ void __launch_bounds__(256, 1)
moe_gemm(const float*  __restrict__ xg,
         const __bf16* __restrict__ w13P,
         const __bf16* __restrict__ w2P,
         const int*    __restrict__ cnt,
         const int*    __restrict__ buckets,
         float*        __restrict__ outg) {
  __shared__ int prow[TILE_M];
  // X tile in A-fragment order: [mt(2)][kb(16)][lane(32)][16 bf16]
  __shared__ __align__(32) __bf16 xA[2 * NKB * 32 * 16];            // 32 KB
  // double-buffered packed weight chunks (filled by async DMA)
  __shared__ __align__(32) __bf16 w13B[2][W13P_PER_CHUNK];          // 128 KB
  __shared__ __align__(32) __bf16 w2B [2][W2P_PER_CHUNK];           // 64 KB
  __shared__ __align__(32) __bf16 hA[2 * 32 * 16];                  // 2 KB
  __shared__ float gu[TILE_M * 64];                                 // 8 KB

  const int tid  = threadIdx.x;
  const int wv   = tid >> 5;       // wave id 0..7 (wave32)
  const int lane = tid & 31;

  const int e    = blockIdx.x / MAX_TILES;
  const int tile = blockIdx.x % MAX_TILES;
  const int nc   = cnt[e];
  const int m0   = tile * TILE_M;
  if (m0 >= nc) return;            // block-uniform exit

  // issue async DMA for chunk 0 weights immediately (overlaps X staging)
  {
    const char* s13 = (const char*)(w13P + ((size_t)e * NCH << 15));
    const char* s2  = (const char*)(w2P  + ((size_t)e * NCH << 14));
    #pragma unroll
    for (int it = 0; it < 16; ++it)
      cp16_to_lds((char*)&w13B[0][0] + it * 4096 + tid * 16, s13 + it * 4096 + tid * 16);
    #pragma unroll
    for (int it = 0; it < 8; ++it)
      cp16_to_lds((char*)&w2B[0][0] + it * 4096 + tid * 16, s2 + it * 4096 + tid * 16);
  }

  if (tid < TILE_M) {
    int g = m0 + tid;
    prow[tid] = (g < nc) ? buckets[e * NPAIR + g] : -1;
  }
  __syncthreads();

  // ---- stage gathered X rows -> bf16 A-fragments (coalesced along H) ----
  #pragma unroll 4
  for (int it = 0; it < (TILE_M * Hdim) / 256; ++it) {  // 64 iters
    int linear = it * 256 + tid;
    int r = linear >> 9;            // row 0..31
    int k = linear & (Hdim - 1);    // col 0..511
    int pair = prow[r];
    int tok  = (pair < 0) ? 0 : (pair / TOPK);
    float v = xg[(size_t)tok * Hdim + k];
    int mt = r >> 4, lr = r & 15;
    int kb = k >> 5, kk = k & 31;
    xA[((mt * NKB + kb) * 32 + (a_half(kk) * 16 + lr)) * 16 + a_j(kk)] = (__bf16)v;
  }

  // fp32 accumulators: wave wv owns (mt2, 8 col tiles)
  v8f acc[8];
  #pragma unroll
  for (int i = 0; i < 8; ++i)
    #pragma unroll
    for (int j = 0; j < 8; ++j) acc[i][j] = 0.0f;

  const int mt2    = wv & 1;          // stage-2 row tile
  const int ntBase = (wv >> 1) * 8;   // stage-2 col tile group
  const int mt1  = wv & 1;            // stage-1 assignment
  const int nt1  = (wv >> 1) & 1;
  const int isUp = (wv >> 2) & 1;
  const int bsel = isUp * 2 + nt1;    // 0,1 = gate cols, 2,3 = up cols

  for (int ci = 0; ci < NCH; ++ci) {
    const int cur = ci & 1;
    const int nxt = cur ^ 1;
    __syncthreads();  // all waves done reading buf[nxt] (chunk ci-1) and gu/hA

    // issue DMA for chunk ci+1 into the other buffer (24 async instrs / wave)
    if (ci + 1 < NCH) {
      const char* s13 = (const char*)(w13P + (((size_t)e * NCH + ci + 1) << 15));
      const char* s2  = (const char*)(w2P  + (((size_t)e * NCH + ci + 1) << 14));
      #pragma unroll
      for (int it = 0; it < 16; ++it)
        cp16_to_lds((char*)&w13B[nxt][0] + it * 4096 + tid * 16, s13 + it * 4096 + tid * 16);
      #pragma unroll
      for (int it = 0; it < 8; ++it)
        cp16_to_lds((char*)&w2B[nxt][0] + it * 4096 + tid * 16, s2 + it * 4096 + tid * 16);
      WAIT_ASYNC(24);   // batch ci complete (in-order); batch ci+1 stays in flight
    } else {
      WAIT_ASYNC(0);    // final chunk: everything landed
    }
    __syncthreads();    // all waves' waits done -> buf[cur] fully valid

    // ---- stage 1: gate/up tile per wave, reduce over H (16 x K=32 WMMAs) ----
    {
      v8f cg;
      #pragma unroll
      for (int j = 0; j < 8; ++j) cg[j] = 0.0f;
      #pragma unroll
      for (int kb = 0; kb < NKB; ++kb) {
        v16bf a = *reinterpret_cast<const v16bf*>(&xA[((mt1 * NKB + kb) * 32 + lane) * 16]);
        v16bf b = *reinterpret_cast<const v16bf*>(&w13B[cur][((kb * 4 + bsel) * 32 + lane) * 16]);
        cg = __builtin_amdgcn_wmma_f32_16x16x32_bf16(false, a, false, b, (short)0, cg,
                                                     false, false);
      }
      // C layout: lane holds N = lane&15; VGPR v holds M = v + 8*(lane>>4)
      int n = lane & 15, hl = lane >> 4;
      #pragma unroll
      for (int v2 = 0; v2 < 8; ++v2) {
        int m = v2 + 8 * hl;
        gu[(mt1 * 16 + m) * 64 + isUp * 32 + nt1 * 16 + n] = cg[v2];
      }
    }
    __syncthreads();

    // ---- SwiGLU in fp32, emit bf16 A-fragments for stage 2 ----
    #pragma unroll
    for (int i = 0; i < 4; ++i) {
      int id = i * 256 + tid;          // 0..1023
      int m = id >> 5, kk = id & 31;
      float g = gu[m * 64 + kk];
      float u = gu[m * 64 + 32 + kk];
      float h = (g / (1.0f + __expf(-g))) * u;   // silu(g)*u
      int mt = m >> 4, lr = m & 15;
      hA[(mt * 32 + (a_half(kk) * 16 + lr)) * 16 + a_j(kk)] = (__bf16)h;
    }
    __syncthreads();

    // ---- stage 2: out += h_chunk (32xK=32) @ w2_chunk (32x512) ----
    {
      v16bf a = *reinterpret_cast<const v16bf*>(&hA[(mt2 * 32 + lane) * 16]);
      #pragma unroll
      for (int i = 0; i < 8; ++i) {
        int nt = ntBase + i;
        v16bf b = *reinterpret_cast<const v16bf*>(&w2B[cur][(nt * 32 + lane) * 16]);
        acc[i] = __builtin_amdgcn_wmma_f32_16x16x32_bf16(false, a, false, b, (short)0,
                                                         acc[i], false, false);
      }
    }
  }

  // ---- scatter accumulators to out rows (out is (B,S,K,H) == (NPAIR, H)) ----
  {
    int n0 = lane & 15, hl = lane >> 4;
    #pragma unroll
    for (int v2 = 0; v2 < 8; ++v2) {
      int m = mt2 * 16 + v2 + 8 * hl;
      int pair = prow[m];
      if (pair >= 0) {
        float* orow = outg + (size_t)pair * Hdim;
        #pragma unroll
        for (int i = 0; i < 8; ++i) {
          orow[(ntBase + i) * 16 + n0] = acc[i][v2];
        }
      }
    }
  }
}

// ---------------- host launch ----------------
extern "C" void kernel_launch(void* const* d_in, const int* in_sizes, int n_in,
                              void* d_out, int out_size, void* d_ws, size_t ws_size,
                              hipStream_t stream) {
  const float* x    = (const float*)d_in[0];   // (B,S,H)
  const int*   eidx = (const int*)  d_in[1];   // (B,S,K)
  const float* w13  = (const float*)d_in[2];   // (E,H,2I)
  const float* w2   = (const float*)d_in[3];   // (E,I,H)
  float*       out  = (float*)d_out;           // (B,S,K,H)

  // workspace layout (≈24.6 MB): counters | buckets | packed bf16 weights
  char* base = (char*)d_ws;
  int* cnt     = (int*)base;                                  // Edim ints (256 B reserved)
  int* buckets = (int*)(base + 256);                          // E*NPAIR ints (512 KB)
  __bf16* w13P = (__bf16*)(base + 256 + (size_t)Edim * NPAIR * 4);
  __bf16* w2P  = w13P + W13P_TOTAL;

  moe_zero_cnt<<<1, 32, 0, stream>>>(cnt);
  moe_scatter<<<NPAIR / 256, 256, 0, stream>>>(eidx, cnt, buckets);
  moe_pack13<<<W13P_TOTAL / (256 * 8), 256, 0, stream>>>(w13, w13P);
  moe_pack2<<<W2P_TOTAL / (256 * 8), 256, 0, stream>>>(w2, w2P);
  moe_gemm<<<Edim * MAX_TILES, 256, 0, stream>>>(x, w13P, w2P, cnt, buckets, out);
}
